// MultiheadSelfAttention_23072564314257
// MI455X (gfx1250) — matrix operand order repaired and checked by reading, not compiled, not run on hardware
//
#include <hip/hip_runtime.h>

// MI455X / gfx1250, wave32. All matmuls via v_wmma_f32_16x16x32_bf16.
// Fragment layouts per CDNA5 ISA 7.12.2.

typedef __bf16 bf16_t;
typedef __bf16 v8bf  __attribute__((ext_vector_type(8)));
typedef __bf16 v16bf __attribute__((ext_vector_type(16)));
typedef float  v8f   __attribute__((ext_vector_type(8)));

#define CDIM   256
#define HW     16384   // 128*128 spatial positions per channel
#define WPIX   128
#define QSCALE 0.42044820762685725f  // 32^(-1/4)

static __device__ __forceinline__ v8f wmma_bf16(v16bf a, v16bf b, v8f c) {
  return __builtin_amdgcn_wmma_f32_16x16x32_bf16(false, a, false, b, (short)0, c,
                                                 false, false);
}

static __device__ __forceinline__ v8f zero8() {
  v8f z;
#pragma unroll
  for (int i = 0; i < 8; ++i) z[i] = 0.0f;
  return z;
}

// ---------- fragment loaders ----------
// A-fragment from pre-swizzled bf16 weights: per row, each 32-wide k block stored in
// natural order 0..31; lane takes 16 contiguous bf16 (32B load).
static __device__ __forceinline__ v16bf a_frag_swz(const bf16_t* W, int row0, int k0,
                                                   int lane) {
  const bf16_t* p = W + (size_t)(row0 + (lane & 15)) * CDIM + k0 + ((lane >> 4) << 4);
  return *(const v16bf*)p;
}

// A-fragment from row-major bf16 (two 16B loads).
static __device__ __forceinline__ v16bf a_frag_bf16(const bf16_t* base, int ld,
                                                    int row0, int k0, int lane) {
  const bf16_t* p = base + (size_t)(row0 + (lane & 15)) * ld + k0 + ((lane >> 4) << 4);
  v8bf lo = *(const v8bf*)p;
  v8bf hi = *(const v8bf*)(p + 8);
  v16bf a;
#pragma unroll
  for (int i = 0; i < 8; ++i) { a[i] = lo[i]; a[i + 8] = hi[i]; }
  return a;
}

// B-fragment where B = M^T, M row-major [N x K]: per-lane contiguous 32B load.
static __device__ __forceinline__ v16bf b_frag_transposed(const bf16_t* base, int ld,
                                                          int n0, int k0, int lane) {
  const bf16_t* p = base + (size_t)(n0 + (lane & 15)) * ld + k0 + ((lane >> 4) << 4);
  return *(const v16bf*)p;
}

// B-fragment from LDS swizzled tile: element (k,n) stored at ((k>>4)*128+n)*16+(k&15).
// Per-lane fragment = 16 contiguous bf16 (2x ds_load_b128).
static __device__ __forceinline__ v16bf b_frag_swz(const bf16_t* buf, int k0, int n0,
                                                   int lane) {
  const bf16_t* p =
      buf + (size_t)(((k0 >> 4) + (lane >> 4)) * 128 + n0 + (lane & 15)) * 16;
  return *(const v16bf*)p;
}

// =============== Kernel 0: swizzle all 4 weight matrices to bf16 A-frag order ========
__global__ void weight_swizzle_kernel(const float* __restrict__ wq,
                                      const float* __restrict__ wk,
                                      const float* __restrict__ wv,
                                      const float* __restrict__ wo,
                                      bf16_t* __restrict__ wz) {
  int idx = blockIdx.x * 256 + threadIdx.x;  // 4*65536 total
  int m   = idx >> 16;
  int r   = idx & 65535;
  int row = r >> 8;
  int col = r & 255;
  int k0  = col & ~31;
  int j   = col & 31;
  int src = j;
  const float* W = (m == 0) ? wq : (m == 1) ? wk : (m == 2) ? wv : wo;
  wz[idx] = (bf16_t)W[row * CDIM + k0 + src];
}

// =============== Kernel 1: fused Q/K/V projection ===============
// grid (128 p-tiles, 8 batches), block 256 (8 waves).
__global__ void qkv_proj_kernel(const float* __restrict__ x,
                                const bf16_t* __restrict__ wz,  // [3][256][256] swizzled
                                const float* __restrict__ bq, const float* __restrict__ bk,
                                const float* __restrict__ bv,
                                bf16_t* __restrict__ q, bf16_t* __restrict__ k,
                                bf16_t* __restrict__ v) {
  extern __shared__ bf16_t ldsX[];  // 32768 elems (64KB), swizzled B-frag layout
  const int tid = threadIdx.x, lane = tid & 31, wave = tid >> 5;
  const int p0 = blockIdx.x * 128, b = blockIdx.y;
  const float* xb = x + (size_t)b * CDIM * HW;

  // stage x tile -> LDS bf16 in swizzled layout (coalesced float2 reads)
  for (int i = tid; i < 256 * 64; i += 256) {
    int c  = i >> 6;
    int pp = (i & 63) * 2;
    float2 f2 = *(const float2*)(xb + (size_t)c * HW + p0 + pp);
    int base = ((c >> 4) * 128 + pp) * 16 + (c & 15);
    ldsX[base]      = (bf16_t)f2.x;
    ldsX[base + 16] = (bf16_t)f2.y;
  }
  __syncthreads();

  const int mbase = (lane >> 4) << 3;

#pragma unroll 1
  for (int kind = 0; kind < 3; ++kind) {
    const bf16_t* W    = wz + (size_t)kind * 65536;
    const float*  bias = (kind == 0) ? bq : (kind == 1) ? bk : bv;
    bf16_t* dstb       = (kind == 0) ? q : (kind == 1) ? k : v;
    bf16_t* dst        = dstb + (size_t)b * CDIM * HW;
    const float scale  = (kind == 0) ? QSCALE : 1.0f;

#pragma unroll 1
    for (int oband = 0; oband < 2; ++oband) {
      const int o0 = (oband * 8 + wave) * 16;
      v8f acc[8];
#pragma unroll
      for (int t = 0; t < 8; ++t) acc[t] = zero8();

      v16bf afr = a_frag_swz(W, o0, 0, lane);  // prefetched A fragment
#pragma unroll 1
      for (int k0 = 0; k0 < CDIM; k0 += 32) {
        v16bf cur = afr;
        int k0n = (k0 + 32 < CDIM) ? (k0 + 32) : k0;  // clamped prefetch
        afr = a_frag_swz(W, o0, k0n, lane);
#pragma unroll
        for (int t = 0; t < 8; ++t)
          acc[t] = wmma_bf16(cur, b_frag_swz(ldsX, k0, t * 16, lane), acc[t]);
      }

      float bias8[8];
#pragma unroll
      for (int r = 0; r < 8; ++r) bias8[r] = bias[o0 + mbase + r];
#pragma unroll
      for (int t = 0; t < 8; ++t) {
#pragma unroll
        for (int r = 0; r < 8; ++r) {
          float val = (acc[t][r] + bias8[r]) * scale;
          dst[(size_t)(o0 + mbase + r) * HW + p0 + t * 16 + (lane & 15)] = (bf16_t)val;
        }
      }
    }
  }
}

// =============== Kernel 2: per-(b,c) attention ===============
// grid 2048 (one slice each), block 256 (8 waves); wave owns 16 rows of H.
__global__ void attn_kernel(const bf16_t* __restrict__ q, const bf16_t* __restrict__ k,
                            const bf16_t* __restrict__ v, bf16_t* __restrict__ att) {
  extern __shared__ bf16_t lds2[];  // [0,16384): V swizzled; [16384,32768): P tiles
  bf16_t* ldsV = lds2;
  bf16_t* Pall = lds2 + 16384;
  const int tid = threadIdx.x, lane = tid & 31, wave = tid >> 5;
  const size_t s = blockIdx.x;
  const bf16_t* qs = q + s * (size_t)HW;
  const bf16_t* ks = k + s * (size_t)HW;
  const bf16_t* vs = v + s * (size_t)HW;
  bf16_t*       os = att + s * (size_t)HW;
  const int r0  = wave * 16;
  const int hi8 = (lane >> 4) << 3;

  // stage V (128x128 bf16) into swizzled LDS layout (coalesced u32 reads)
  {
    uint16_t* l16 = (uint16_t*)ldsV;
    for (int i = tid; i < 128 * 64; i += 256) {
      int y  = i >> 6;
      int ww = (i & 63) * 2;
      uint32_t u = *(const uint32_t*)(vs + (size_t)y * WPIX + ww);
      int base = ((y >> 4) * 128 + ww) * 16 + (y & 15);
      l16[base]      = (uint16_t)u;
      l16[base + 16] = (uint16_t)(u >> 16);
    }
  }
  __syncthreads();

  // ---- S = Q_band(16x128) * K^T, in registers ----
  v8f acc[8];
#pragma unroll
  for (int t = 0; t < 8; ++t) acc[t] = zero8();
  v16bf qfr = a_frag_bf16(qs, WPIX, r0, 0, lane);
#pragma unroll 1
  for (int k0 = 0; k0 < WPIX; k0 += 32) {
    v16bf cur = qfr;
    int k0n = (k0 + 32 < WPIX) ? (k0 + 32) : k0;
    qfr = a_frag_bf16(qs, WPIX, r0, k0n, lane);
#pragma unroll
    for (int t = 0; t < 8; ++t)
      acc[t] = wmma_bf16(cur, b_frag_transposed(ks, WPIX, t * 16, k0, lane), acc[t]);
  }

  // ---- softmax over y (row m lives in elem r across one 16-lane half) ----
  bf16_t* pb = Pall + wave * 2048;
#pragma unroll
  for (int r = 0; r < 8; ++r) {
    float m = acc[0][r];
#pragma unroll
    for (int t = 1; t < 8; ++t) m = fmaxf(m, acc[t][r]);
#pragma unroll
    for (int off = 1; off < 16; off <<= 1) m = fmaxf(m, __shfl_xor(m, off, 32));
    float e[8], sum = 0.0f;
#pragma unroll
    for (int t = 0; t < 8; ++t) { e[t] = __expf(acc[t][r] - m); sum += e[t]; }
#pragma unroll
    for (int off = 1; off < 16; off <<= 1) sum += __shfl_xor(sum, off, 32);
    float rinv = __frcp_rn(sum);
#pragma unroll
    for (int t = 0; t < 8; ++t)
      pb[(r + hi8) * WPIX + t * 16 + (lane & 15)] = (bf16_t)(e[t] * rinv);
  }
  // pb is wave-private; same-wave DS ops are in-order, no block barrier needed.

  // ---- O = P(16x128) * V(128x128), V B-frags from swizzled LDS ----
#pragma unroll 1
  for (int t = 0; t < 8; ++t) {
    v8f oacc = zero8();
#pragma unroll
    for (int k0 = 0; k0 < WPIX; k0 += 32) {
      v16bf afr = a_frag_bf16(pb, WPIX, 0, k0, lane);
      oacc = wmma_bf16(afr, b_frag_swz(ldsV, k0, t * 16, lane), oacc);
    }
#pragma unroll
    for (int r = 0; r < 8; ++r)
      os[(size_t)(r0 + hi8 + r) * WPIX + t * 16 + (lane & 15)] = (bf16_t)oacc[r];
  }
}

// =============== Kernel 3: output projection, fp32 out ===============
__global__ void out_proj_kernel(const bf16_t* __restrict__ att,
                                const bf16_t* __restrict__ wzo,  // swizzled wo
                                const float* __restrict__ bo, float* __restrict__ out) {
  extern __shared__ bf16_t ldsX[];  // 32768 elems, swizzled
  const int tid = threadIdx.x, lane = tid & 31, wave = tid >> 5;
  const int p0 = blockIdx.x * 128, b = blockIdx.y;
  const bf16_t* src = att + (size_t)b * CDIM * HW;

  {
    uint16_t* l16 = (uint16_t*)ldsX;
    for (int i = tid; i < 256 * 64; i += 256) {
      int c  = i >> 6;
      int pp = (i & 63) * 2;
      uint32_t u = *(const uint32_t*)(src + (size_t)c * HW + p0 + pp);
      int base = ((c >> 4) * 128 + pp) * 16 + (c & 15);
      l16[base]      = (uint16_t)u;
      l16[base + 16] = (uint16_t)(u >> 16);
    }
  }
  __syncthreads();

  float* outp = out + (size_t)b * CDIM * HW;
  const int mbase = (lane >> 4) << 3;
#pragma unroll 1
  for (int oband = 0; oband < 2; ++oband) {
    const int o0 = (oband * 8 + wave) * 16;
    v8f acc[8];
#pragma unroll
    for (int t = 0; t < 8; ++t) acc[t] = zero8();

    v16bf afr = a_frag_swz(wzo, o0, 0, lane);
#pragma unroll 1
    for (int k0 = 0; k0 < CDIM; k0 += 32) {
      v16bf cur = afr;
      int k0n = (k0 + 32 < CDIM) ? (k0 + 32) : k0;
      afr = a_frag_swz(wzo, o0, k0n, lane);
#pragma unroll
      for (int t = 0; t < 8; ++t)
        acc[t] = wmma_bf16(cur, b_frag_swz(ldsX, k0, t * 16, lane), acc[t]);
    }
    float bias8[8];
#pragma unroll
    for (int r = 0; r < 8; ++r) bias8[r] = bo[o0 + mbase + r];
#pragma unroll
    for (int t = 0; t < 8; ++t) {
#pragma unroll
      for (int r = 0; r < 8; ++r)
        outp[(size_t)(o0 + mbase + r) * HW + p0 + t * 16 + (lane & 15)] =
            acc[t][r] + bias8[r];
    }
  }
}

// =============== host launch ===============
extern "C" void kernel_launch(void* const* d_in, const int* in_sizes, int n_in,
                              void* d_out, int out_size, void* d_ws, size_t ws_size,
                              hipStream_t stream) {
  const float* x  = (const float*)d_in[0];
  const float* wq = (const float*)d_in[1];
  const float* bq = (const float*)d_in[2];
  const float* wk = (const float*)d_in[3];
  const float* bk = (const float*)d_in[4];
  const float* wv = (const float*)d_in[5];
  const float* bv = (const float*)d_in[6];
  const float* wo = (const float*)d_in[7];
  const float* bo = (const float*)d_in[8];
  float* out = (float*)d_out;

  // workspace: q,k,v,att bf16 (4 x 64MB) + swizzled weights (512KB)
  const size_t NEL = (size_t)8 * CDIM * HW;  // 33,554,432
  bf16_t* q   = (bf16_t*)d_ws;
  bf16_t* k   = q + NEL;
  bf16_t* v   = k + NEL;
  bf16_t* att = v + NEL;
  bf16_t* wz  = att + NEL;  // [4][256][256]

  const size_t lds_bytes = 32768 * sizeof(bf16_t);  // 64 KB

  weight_swizzle_kernel<<<dim3(1024), dim3(256), 0, stream>>>(wq, wk, wv, wo, wz);
  qkv_proj_kernel<<<dim3(128, 8), dim3(256), lds_bytes, stream>>>(
      x, wz, bq, bk, bv, q, k, v);
  attn_kernel<<<dim3(2048), dim3(256), lds_bytes, stream>>>(q, k, v, att);
  out_proj_kernel<<<dim3(128, 8), dim3(256), lds_bytes, stream>>>(
      att, wz + (size_t)3 * 65536, bo, out);
}
